// LayerScaleBlock_27805618275035
// MI455X (gfx1250) — matrix operand order repaired and checked by reading, not compiled, not run on hardware
//
#include <hip/hip_runtime.h>
#include <hip/hip_bf16.h>
#include <stdint.h>

// ---------------------------------------------------------------------------
// LayerScaleBlock (talking-heads attention + MLP) for MI455X / gfx1250.
// All matrix math on v_wmma_f32_16x16x32_f16 (wave32 WMMA).
// ---------------------------------------------------------------------------

typedef __attribute__((ext_vector_type(16))) _Float16 v16h;
typedef __attribute__((ext_vector_type(8)))  float    v8f;

#define B_  8
#define N_  1024
#define D_  384
#define H_  8
#define HD_ 48
#define HDP 64          // padded head dim for exact K=32 WMMA steps
#define HID_ 1536
#define ROWS (B_ * N_)  // 8192

__device__ __forceinline__ v8f wmma_f16(v16h a, v16h b, v8f c) {
  return __builtin_amdgcn_wmma_f32_16x16x32_f16(false, a, false, b, (short)0, c,
                                                false, false);
}

// A-matrix fragment: 16x32 f16, row-major source, leading dim `ld`.
// Layout (ISA 7.12.2): lanes 0-15 row=lane, halves0-7 = K0..7, halves8-15 =
// K16..23 ; lanes 16-31 row=lane-16, halves = K8..15 / K24..31.
__device__ __forceinline__ v16h load_a16x32(const _Float16* __restrict__ p, int ld) {
  const int lane = threadIdx.x & 31;
  const int row  = lane & 15;
  const int kb   = (lane < 16) ? 0 : 8;
  const _Float16* r0 = p + (size_t)row * ld + kb;
  v16h a;
#pragma unroll
  for (int i = 0; i < 8; ++i) { a[i] = r0[i]; a[i + 8] = r0[16 + i]; }
  return a;
}

// B-matrix fragment (32x16, K x N) where B = W^T and W is [n][k] row-major.
// p = W + n0*ld + k0.  Lanes 0-15: col=lane, halves = K0..7 / K16..23; lanes
// 16-31: col=lane-16, halves = K8..15 / K24..31.
__device__ __forceinline__ v16h load_bT(const _Float16* __restrict__ p, int ld) {
  const int lane = threadIdx.x & 31;
  const int col  = lane & 15;
  const int kb   = (lane < 16) ? 0 : 8;
  const _Float16* r0 = p + (size_t)col * ld + kb;
  v16h b;
#pragma unroll
  for (int i = 0; i < 8; ++i) { b[i] = r0[i]; b[i + 8] = r0[16 + i]; }
  return b;
}

// B-matrix fragment from a K-major (row-major [k][n]) source; p = Bm + k0*ld + n0.
__device__ __forceinline__ v16h load_bK(const _Float16* __restrict__ p, int ld) {
  const int lane = threadIdx.x & 31;
  const int col  = lane & 15;
  const int kb   = (lane < 16) ? 0 : 8;
  v16h b;
#pragma unroll
  for (int i = 0; i < 16; ++i) b[i] = p[(size_t)(kb + (i & 7) + (i >> 3) * 16) * ld + col];
  return b;
}

// ---------------------------------------------------------------------------
// Utility kernels
// ---------------------------------------------------------------------------
__global__ void cvt_f32_to_f16(const float* __restrict__ src,
                               _Float16* __restrict__ dst, int n) {
  for (int i = blockIdx.x * blockDim.x + threadIdx.x; i < n;
       i += gridDim.x * blockDim.x)
    dst[i] = (_Float16)src[i];
}

__global__ void zero_u32(uint32_t* __restrict__ p, int n) {
  for (int i = blockIdx.x * blockDim.x + threadIdx.x; i < n;
       i += gridDim.x * blockDim.x)
    p[i] = 0u;
}

// LayerNorm: one wave per 384-element row, f16 output.
__global__ __launch_bounds__(256) void ln_kernel(const float* __restrict__ x,
                                                 const float* __restrict__ g,
                                                 const float* __restrict__ b,
                                                 _Float16* __restrict__ out) {
  const int wave = threadIdx.x >> 5, lane = threadIdx.x & 31;
  const int row = blockIdx.x * 8 + wave;
  const float* xr = x + (size_t)row * D_;
  float v[12], s = 0.f, sq = 0.f;
#pragma unroll
  for (int i = 0; i < 12; ++i) {
    v[i] = xr[lane + i * 32];
    s += v[i];
    sq += v[i] * v[i];
  }
#pragma unroll
  for (int sh = 16; sh >= 1; sh >>= 1) {
    s  += __shfl_xor(s, sh, 32);
    sq += __shfl_xor(sq, sh, 32);
  }
  const float mean = s * (1.f / D_);
  const float var  = sq * (1.f / D_) - mean * mean;
  const float rstd = rsqrtf(var + 1e-5f);
  _Float16* orow = out + (size_t)row * D_;
#pragma unroll
  for (int i = 0; i < 12; ++i) {
    const int c = lane + i * 32;
    orow[c] = (_Float16)((v[i] - mean) * rstd * g[c] + b[c]);
  }
}

// ---------------------------------------------------------------------------
// Generic WMMA GEMM: C[M x N] = A[M x K](f16) * W[N x K]^T(f16).
// Wave tile 32x64: 2 A-fragments x 4 B-fragments -> 8 WMMAs per k-step,
// each B-fragment reused twice.  Block = 8 waves -> 256(M) x 64(N) tile.
// MODE 0: scatter into padded per-head qkv buffer (scale q)
// MODE 1: outF = resid + gamma * (c + bias)          (attention proj)
// MODE 2: outH = gelu_exact(c + bias)                (fc1)
// MODE 3: outF = resid + gamma * (c + bias)          (fc2 -> d_out)
// ---------------------------------------------------------------------------
template <int MODE>
__global__ __launch_bounds__(256) void gemm_kernel(
    const _Float16* __restrict__ A, const _Float16* __restrict__ W,
    const float* __restrict__ bias, const float* __restrict__ gamma,
    const float* __restrict__ resid, float* __restrict__ outF,
    _Float16* __restrict__ outH, int M, int N, int K) {
  const int wave = threadIdx.x >> 5, lane = threadIdx.x & 31;
  const int m0 = blockIdx.y * 256 + wave * 32;
  const int n0 = blockIdx.x * 64;
  const int colL  = lane & 15;
  const int rbase = (lane < 16) ? 0 : 8;

  v8f acc[2][4] = {{v8f{}, v8f{}, v8f{}, v8f{}}, {v8f{}, v8f{}, v8f{}, v8f{}}};
  const _Float16* Ap0 = A + (size_t)m0 * K;
  const _Float16* Ap1 = A + (size_t)(m0 + 16) * K;
  for (int k0 = 0; k0 < K; k0 += 32) {
    const v16h a0 = load_a16x32(Ap0 + k0, K);
    const v16h a1 = load_a16x32(Ap1 + k0, K);
    if (k0 + 32 < K) {  // hide next-step latency (global_prefetch_b8)
      __builtin_prefetch(Ap0 + ((size_t)(lane & 15)) * K + k0 + 32, 0, 1);
      __builtin_prefetch(Ap1 + ((size_t)(lane & 15)) * K + k0 + 32, 0, 1);
    }
#pragma unroll
    for (int t = 0; t < 4; ++t) {
      const v16h b = load_bT(W + (size_t)(n0 + t * 16) * K + k0, K);
      acc[0][t] = wmma_f16(a0, b, acc[0][t]);
      acc[1][t] = wmma_f16(a1, b, acc[1][t]);
    }
  }

  constexpr float QSCALE = 0.14433756729740643f;  // 48^-0.5
#pragma unroll
  for (int mt = 0; mt < 2; ++mt) {
#pragma unroll
    for (int t = 0; t < 4; ++t) {
#pragma unroll
      for (int r = 0; r < 8; ++r) {
        const int n = n0 + t * 16 + colL;
        const int m = m0 + mt * 16 + rbase + r;
        float c = acc[mt][t][r];
        if constexpr (MODE == 0) {
          // n in [0,1152): t3 = q/k/v, head, hd.  m = b*1024 + s.
          const int t3   = n / D_;
          const int rem  = n - t3 * D_;
          const int head = rem / HD_;
          const int hd   = rem - head * HD_;
          const int bi   = m >> 10, s = m & (N_ - 1);
          if (t3 == 0) c *= QSCALE;
          const size_t idx =
              (((size_t)(t3 * B_ + bi) * H_ + head) * N_ + s) * HDP + hd;
          outH[idx] = (_Float16)c;
        } else if constexpr (MODE == 2) {
          c += bias[n];
          const float ge = 0.5f * c * (1.0f + erff(c * 0.70710678118654752f));
          outH[(size_t)m * HID_ + n] = (_Float16)ge;
        } else {  // MODE 1 / 3
          c += bias[n];
          outF[(size_t)m * D_ + n] =
              resid[(size_t)m * D_ + n] + gamma[n] * c;
        }
      }
    }
  }
}

// ---------------------------------------------------------------------------
// Talking-heads attention.  Grid: (N/16, B); block 256 = 8 waves, wave w owns
// head w.  Two passes over j (pre-softmax mixing prevents online factoring):
//   pass1: raw scores (WMMA) -> LDS exchange -> w_l mix -> row max/sum
//   pass2: recompute raw -> mix -> normalized P -> LDS exchange -> w_w mix
//          assembled directly as f16 A-fragment -> WMMA against V.
// ---------------------------------------------------------------------------
__global__ __launch_bounds__(256) void attn_kernel(
    const _Float16* __restrict__ qkvp, const float* __restrict__ wl,
    const float* __restrict__ bl, const float* __restrict__ ww,
    const float* __restrict__ bw, _Float16* __restrict__ o16) {
  __shared__ float sbuf[H_][32][17];  // [head][col 0..31][row, padded]

  const int bi = blockIdx.y, it = blockIdx.x;
  const int wave = threadIdx.x >> 5, lane = threadIdx.x & 31;
  const int g = wave;
  const int colL  = lane & 15;
  const int rbase = (lane < 16) ? 0 : 8;
  const int i0 = it * 16;

  const _Float16* Qp = qkvp + ((size_t)((0 * B_ + bi) * H_ + g) * N_) * HDP;
  const _Float16* Kp = qkvp + ((size_t)((1 * B_ + bi) * H_ + g) * N_) * HDP;
  const _Float16* Vp = qkvp + ((size_t)((2 * B_ + bi) * H_ + g) * N_) * HDP;

  const v16h qa0 = load_a16x32(Qp + (size_t)i0 * HDP + 0, HDP);
  const v16h qa1 = load_a16x32(Qp + (size_t)i0 * HDP + 32, HDP);

  float wlr[H_], wwr[H_];
#pragma unroll
  for (int h = 0; h < H_; ++h) {
    wlr[h] = wl[g * H_ + h];
    wwr[h] = ww[g * H_ + h];
  }
  const float blg = bl[g], bwg = bw[g];

  float mrow[8], srow[8];
#pragma unroll
  for (int r = 0; r < 8; ++r) { mrow[r] = -1e30f; srow[r] = 0.f; }

  // ---------------- pass 1: row max / sum of mixed logits ----------------
  for (int j0 = 0; j0 < N_; j0 += 32) {
    if (j0 + 32 < N_)
      __builtin_prefetch(Kp + (size_t)(j0 + 32 + colL) * HDP, 0, 1);
    v8f c0 = {}, c1 = {};
    c0 = wmma_f16(qa0, load_bT(Kp + (size_t)(j0 + 0) * HDP + 0, HDP), c0);
    c0 = wmma_f16(qa1, load_bT(Kp + (size_t)(j0 + 0) * HDP + 32, HDP), c0);
    c1 = wmma_f16(qa0, load_bT(Kp + (size_t)(j0 + 16) * HDP + 0, HDP), c1);
    c1 = wmma_f16(qa1, load_bT(Kp + (size_t)(j0 + 16) * HDP + 32, HDP), c1);
#pragma unroll
    for (int r = 0; r < 8; ++r) {
      sbuf[g][colL][rbase + r]      = c0[r];
      sbuf[g][16 + colL][rbase + r] = c1[r];
    }
    __syncthreads();
#pragma unroll
    for (int r = 0; r < 8; ++r) {
      float l0 = blg, l1 = blg;
#pragma unroll
      for (int h = 0; h < H_; ++h) {
        l0 += wlr[h] * sbuf[h][colL][rbase + r];
        l1 += wlr[h] * sbuf[h][16 + colL][rbase + r];
      }
      float nm = fmaxf(mrow[r], l0);
      srow[r] = srow[r] * __expf(mrow[r] - nm) + __expf(l0 - nm);
      mrow[r] = nm;
      nm = fmaxf(mrow[r], l1);
      srow[r] = srow[r] * __expf(mrow[r] - nm) + __expf(l1 - nm);
      mrow[r] = nm;
    }
    __syncthreads();
  }
  // reduce stats across the 16 lanes of each half-wave (xor < 16 keeps halves)
#pragma unroll
  for (int sh = 1; sh < 16; sh <<= 1) {
#pragma unroll
    for (int r = 0; r < 8; ++r) {
      const float om = __shfl_xor(mrow[r], sh, 32);
      const float os = __shfl_xor(srow[r], sh, 32);
      const float nm = fmaxf(mrow[r], om);
      srow[r] = srow[r] * __expf(mrow[r] - nm) + os * __expf(om - nm);
      mrow[r] = nm;
    }
  }
  float sinv[8];
#pragma unroll
  for (int r = 0; r < 8; ++r) sinv[r] = 1.0f / srow[r];

  // ---------------- pass 2: P -> w_w mix -> WMMA against V ----------------
  v8f oacc[4] = {v8f{}, v8f{}, v8f{}, v8f{}};
  const int kb0 = (lane < 16) ? 0 : 8;  // A-fragment K offsets for this lane
  for (int j0 = 0; j0 < N_; j0 += 32) {
    if (j0 + 32 < N_) {
      __builtin_prefetch(Kp + (size_t)(j0 + 32 + colL) * HDP, 0, 1);
      __builtin_prefetch(Vp + (size_t)(j0 + 32 + colL) * HDP, 0, 1);
    }
    v8f c0 = {}, c1 = {};
    c0 = wmma_f16(qa0, load_bT(Kp + (size_t)(j0 + 0) * HDP + 0, HDP), c0);
    c0 = wmma_f16(qa1, load_bT(Kp + (size_t)(j0 + 0) * HDP + 32, HDP), c0);
    c1 = wmma_f16(qa0, load_bT(Kp + (size_t)(j0 + 16) * HDP + 0, HDP), c1);
    c1 = wmma_f16(qa1, load_bT(Kp + (size_t)(j0 + 16) * HDP + 32, HDP), c1);
#pragma unroll
    for (int r = 0; r < 8; ++r) {
      sbuf[g][colL][rbase + r]      = c0[r];
      sbuf[g][16 + colL][rbase + r] = c1[r];
    }
    __syncthreads();
    float p0[8], p1[8];
#pragma unroll
    for (int r = 0; r < 8; ++r) {
      float l0 = blg, l1 = blg;
#pragma unroll
      for (int h = 0; h < H_; ++h) {
        l0 += wlr[h] * sbuf[h][colL][rbase + r];
        l1 += wlr[h] * sbuf[h][16 + colL][rbase + r];
      }
      p0[r] = __expf(l0 - mrow[r]) * sinv[r];
      p1[r] = __expf(l1 - mrow[r]) * sinv[r];
    }
    __syncthreads();
#pragma unroll
    for (int r = 0; r < 8; ++r) {
      sbuf[g][colL][rbase + r]      = p0[r];
      sbuf[g][16 + colL][rbase + r] = p1[r];
    }
    __syncthreads();
    // assemble mixed-probability A fragment (rows i, K = j within 32-block)
    v16h af;
#pragma unroll
    for (int i = 0; i < 8; ++i) {
      float v0 = bwg, v1 = bwg;
#pragma unroll
      for (int h = 0; h < H_; ++h) {
        v0 += wwr[h] * sbuf[h][kb0 + i][colL];
        v1 += wwr[h] * sbuf[h][kb0 + 16 + i][colL];
      }
      af[i]     = (_Float16)v0;
      af[i + 8] = (_Float16)v1;
    }
#pragma unroll
    for (int t = 0; t < 4; ++t) {
      const v16h bv = load_bK(Vp + (size_t)j0 * HDP + t * 16, HDP);
      oacc[t] = wmma_f16(af, bv, oacc[t]);  // padded V cols are zero
    }
    __syncthreads();
  }

  const int rowg = bi * N_ + i0;
#pragma unroll
  for (int t = 0; t < 4; ++t) {
#pragma unroll
    for (int r = 0; r < 8; ++r) {
      const int col = t * 16 + colL;
      if (col < HD_)
        o16[(size_t)(rowg + rbase + r) * D_ + g * HD_ + col] =
            (_Float16)oacc[t][r];
    }
  }
}

// ---------------------------------------------------------------------------
// Host launcher
// ---------------------------------------------------------------------------
extern "C" void kernel_launch(void* const* d_in, const int* in_sizes, int n_in,
                              void* d_out, int out_size, void* d_ws,
                              size_t ws_size, hipStream_t stream) {
  const float* x      = (const float*)d_in[0];
  const float* ln1_g  = (const float*)d_in[1];
  const float* ln1_b  = (const float*)d_in[2];
  const float* w_qkv  = (const float*)d_in[3];
  const float* w_proj = (const float*)d_in[4];
  const float* b_proj = (const float*)d_in[5];
  const float* w_l    = (const float*)d_in[6];
  const float* b_l    = (const float*)d_in[7];
  const float* w_w    = (const float*)d_in[8];
  const float* b_w    = (const float*)d_in[9];
  const float* ln2_g  = (const float*)d_in[10];
  const float* ln2_b  = (const float*)d_in[11];
  const float* w_fc1  = (const float*)d_in[12];
  const float* b_fc1  = (const float*)d_in[13];
  const float* w_fc2  = (const float*)d_in[14];
  const float* b_fc2  = (const float*)d_in[15];
  const float* gamma1 = (const float*)d_in[16];
  const float* gamma2 = (const float*)d_in[17];

  char* ws = (char*)d_ws;
  size_t off = 0;
  auto take = [&](size_t bytes) {
    char* p = ws + off;
    off = (off + bytes + 255) & ~(size_t)255;
    return p;
  };
  _Float16* wqkv16  = (_Float16*)take((size_t)3 * D_ * D_ * 2);     // [1152][384]
  _Float16* wproj16 = (_Float16*)take((size_t)D_ * D_ * 2);
  _Float16* wfc116  = (_Float16*)take((size_t)HID_ * D_ * 2);
  _Float16* wfc216  = (_Float16*)take((size_t)D_ * HID_ * 2);
  _Float16* h16     = (_Float16*)take((size_t)ROWS * D_ * 2);       // LN out (reused)
  _Float16* o16     = (_Float16*)take((size_t)ROWS * D_ * 2);       // attn out
  float*    x1      = (float*)take((size_t)ROWS * D_ * 4);          // x + g1*proj
  _Float16* qkvp    = (_Float16*)take((size_t)3 * B_ * H_ * N_ * HDP * 2);
  _Float16* gelu16  = qkvp;  // reused after attention completes (same stream)

  // weights -> f16
  cvt_f32_to_f16<<<512, 256, 0, stream>>>(w_qkv, wqkv16, 3 * D_ * D_);
  cvt_f32_to_f16<<<256, 256, 0, stream>>>(w_proj, wproj16, D_ * D_);
  cvt_f32_to_f16<<<512, 256, 0, stream>>>(w_fc1, wfc116, HID_ * D_);
  cvt_f32_to_f16<<<512, 256, 0, stream>>>(w_fc2, wfc216, D_ * HID_);
  // zero padded qkv buffer (pads 48..63 must be 0)
  zero_u32<<<2048, 256, 0, stream>>>((uint32_t*)qkvp,
                                     3 * B_ * H_ * N_ * HDP / 2);

  // LN1 -> h16
  ln_kernel<<<ROWS / 8, 256, 0, stream>>>(x, ln1_g, ln1_b, h16);
  // QKV GEMM -> padded per-head buffer (q scaled)
  gemm_kernel<0><<<dim3((3 * D_) / 64, ROWS / 256), 256, 0, stream>>>(
      h16, wqkv16, nullptr, nullptr, nullptr, nullptr, qkvp, ROWS, 3 * D_, D_);
  // talking-heads attention -> o16
  attn_kernel<<<dim3(N_ / 16, B_), 256, 0, stream>>>(qkvp, w_l, b_l, w_w, b_w,
                                                     o16);
  // proj + layer-scale residual -> x1
  gemm_kernel<1><<<dim3(D_ / 64, ROWS / 256), 256, 0, stream>>>(
      o16, wproj16, b_proj, gamma1, x, x1, nullptr, ROWS, D_, D_);
  // LN2 -> h16 (reuse)
  ln_kernel<<<ROWS / 8, 256, 0, stream>>>(x1, ln2_g, ln2_b, h16);
  // fc1 + exact GELU -> gelu16
  gemm_kernel<2><<<dim3(HID_ / 64, ROWS / 256), 256, 0, stream>>>(
      h16, wfc116, b_fc1, nullptr, nullptr, nullptr, gelu16, ROWS, HID_, D_);
  // fc2 + layer-scale residual -> d_out
  gemm_kernel<3><<<dim3(D_ / 64, ROWS / 256), 256, 0, stream>>>(
      gelu16, wfc216, b_fc2, gamma2, x1, (float*)d_out, nullptr, ROWS, D_,
      HID_);
}
